// model_network_custom_weights_9148280340625
// MI455X (gfx1250) — matrix operand order repaired and checked by reading, not compiled, not run on hardware
//
#include <hip/hip_runtime.h>

// ---------------------------------------------------------------- types
typedef __bf16 bf16_t;
typedef __attribute__((ext_vector_type(16))) __bf16 v16bf;
typedef __attribute__((ext_vector_type(8)))  __bf16 v8bf;
typedef __attribute__((ext_vector_type(8)))  float  v8f;

// ---------------------------------------------------------------- problem dims
#define B_N    1300
#define T_N    128
#define S_N    12
#define K_N    5
#define H_N    64
#define G3     192          // 3*H
#define NSTEPS 122          // T - 1 - K
#define THRESH 100.0f
#define NBLK   21           // persistent blocks: 21*4 waves = 84 tiles >= ceil(1300/16)
#define NTHR   (NBLK * 128)

// ---------------------------------------------------------------- ws layout (bytes)
#define OFF_SPK   0u                             // 1300*128*12 f32 = 7,987,200 B
#define OFF_MASK  7987200u                       // 1300 f32 (padded)
#define OFF_BAR   7992576u                       // 2 x u32 barrier state (padded)
#define OFF_WHT   7992832u                       // 192*64 bf16 = 24,576 B
#define OFF_WXT   8017408u                       // 192*32 bf16 = 12,288 B

// ---------------------------------------------------------------- helpers
__device__ __forceinline__ v16bf cat8(v8bf a, v8bf b) {
  return __builtin_shufflevector(a, b, 0,1,2,3,4,5,6,7,8,9,10,11,12,13,14,15);
}
__device__ __forceinline__ v8f wmma_bf16(v16bf a, v16bf b, v8f c) {
  // D(16x16,f32) = A(16x32,bf16) x B(32x16,bf16) + C
  return __builtin_amdgcn_wmma_f32_16x16x32_bf16(false, a, false, b, (short)0, c,
                                                 false, false);
}
__device__ __forceinline__ float fast_rcp(float x) { return __builtin_amdgcn_rcpf(x); }
__device__ __forceinline__ float fast_sigmoid(float x) { return fast_rcp(1.0f + __expf(-x)); }
__device__ __forceinline__ float fast_tanh(float x) {
  return 1.0f - 2.0f * fast_rcp(1.0f + __expf(2.0f * x));
}

// device-wide sense-reversing barrier (all NBLK blocks co-resident).
// ACQ_REL arrive orders prior plain stores; ACQUIRE spin/arrive invalidates
// caches so plain loads after the barrier observe remote writes.
__device__ __forceinline__ void grid_barrier(unsigned* cnt, unsigned* gen) {
  __syncthreads();
  if (threadIdx.x == 0) {
    unsigned my_gen = __hip_atomic_load(gen, __ATOMIC_RELAXED, __HIP_MEMORY_SCOPE_AGENT);
    unsigned prev = __hip_atomic_fetch_add(cnt, 1u, __ATOMIC_ACQ_REL, __HIP_MEMORY_SCOPE_AGENT);
    if (prev + 1u == (unsigned)NBLK) {
      // all arrivals for this round done; safe to reset before release
      __hip_atomic_store(cnt, 0u, __ATOMIC_RELAXED, __HIP_MEMORY_SCOPE_AGENT);
      __hip_atomic_fetch_add(gen, 1u, __ATOMIC_ACQ_REL, __HIP_MEMORY_SCOPE_AGENT);
    } else {
      while (__hip_atomic_load(gen, __ATOMIC_ACQUIRE, __HIP_MEMORY_SCOPE_AGENT) == my_gen)
        __builtin_amdgcn_s_sleep(1);
    }
  }
  __syncthreads();
}

// ---------------------------------------------------------------- init: copy spikes, zero state
__global__ void init_kernel(const float* __restrict__ spk_in, float* __restrict__ spk_ws,
                            float* __restrict__ maskv, unsigned* __restrict__ bar) {
  const int stride = gridDim.x * blockDim.x;
  const int tid = blockIdx.x * blockDim.x + threadIdx.x;
  for (int i = tid; i < B_N * T_N * S_N; i += stride) spk_ws[i] = spk_in[i];
  for (int i = tid; i < B_N; i += stride) maskv[i] = 0.0f;
  if (tid == 0) { bar[0] = 0u; bar[1] = 0u; }
}

// ---------------------------------------------------------------- prep: bf16 transposed weights
// whT[n*64 + k] = Wh[k][n]   (B-matrix image, column n contiguous in k)
// wxT[n*32 + k] = Wx[k][n] for k<12 else 0   (K padded 12 -> 32)
__global__ void prep_kernel(const float* __restrict__ Wx, const float* __restrict__ Wh,
                            bf16_t* __restrict__ wxT, bf16_t* __restrict__ whT) {
  const int stride = gridDim.x * blockDim.x;
  const int tid = blockIdx.x * blockDim.x + threadIdx.x;
  for (int i = tid; i < G3 * H_N; i += stride) {
    int n = i / H_N, k = i % H_N;
    whT[i] = (bf16_t)Wh[k * G3 + n];
  }
  for (int i = tid; i < G3 * 32; i += stride) {
    int n = i / 32, k = i % 32;
    float v = (k < S_N) ? Wx[k * G3 + n] : 0.0f;
    wxT[i] = (bf16_t)v;
  }
}

// ---------------------------------------------------------------- persistent kernel: all 122 iterations
// 1 wave = one 16-batch tile; weights live in LDS and h/prev_pred live in VGPRs
// for the entire run. Phase 1 = 5 GRU steps + pred + fire mask; Phase 2 =
// event-sparse feedback with deterministic index compaction.
__global__ __launch_bounds__(128) void snn_persistent_kernel(
    float* __restrict__ spk, float* __restrict__ maskv, unsigned* __restrict__ bar,
    const float* __restrict__ wmat,
    const bf16_t* __restrict__ whT, const bf16_t* __restrict__ wxT,
    const float* __restrict__ bx, const float* __restrict__ bh,
    const float* __restrict__ Wo, const float* __restrict__ bo,
    float* __restrict__ preds)
{
  __shared__ __attribute__((aligned(32))) bf16_t sWhT[G3 * H_N];   // 24 KB
  __shared__ __attribute__((aligned(32))) bf16_t sWxT[G3 * 32];    // 12 KB
  __shared__ __attribute__((aligned(32))) bf16_t hbuf[4][16][H_N]; //  8 KB
  __shared__ __attribute__((aligned(32))) bf16_t xbuf[4][16][32];  //  4 KB
  __shared__ float predbuf[4][16][16];                             //  4 KB
  __shared__ int   s_cnt[128];                                     // 512 B
  __shared__ int   s_list[B_N];                                    // 5.2 KB

  const int tid = threadIdx.x;
  const int w  = tid >> 5;       // wave in block
  const int l  = tid & 31;       // lane
  const int hi = l >> 4;         // lane half
  const int lo = l & 15;
  unsigned* bar_cnt = bar;
  unsigned* bar_gen = bar + 1;

  // stage weights once for all 122 iterations
  for (int i = tid; i < G3 * H_N; i += 128) sWhT[i] = whT[i];
  for (int i = tid; i < G3 * 32;  i += 128) sWxT[i] = wxT[i];
  __syncthreads();

  const int tile = blockIdx.x * 4 + w;
  const int b0   = tile * 16;

  // zero-pad x K-range 12..31 once
  for (int i = l; i < 16 * 20; i += 32) xbuf[w][i / 20][12 + i % 20] = (bf16_t)0.0f;

  // persistent per-lane state: h (C/D layout: elem v <-> batch v+8*hi, j q*16+lo)
  v8f hreg[4];
#pragma unroll
  for (int q = 0; q < 4; ++q)
#pragma unroll
    for (int v = 0; v < 8; ++v) hreg[q][v] = 0.0f;
  float prevPred = 0.0f;       // owned by lanes l<16 (batch b0+l)

  // per-lane biases at N = tt*16 + lo
  float biasR[4], biasZ[4], biasX[4], biasH[4], wo0[4];
#pragma unroll
  for (int q = 0; q < 4; ++q) {
    biasR[q] = bx[q * 16 + lo]      + bh[q * 16 + lo];
    biasZ[q] = bx[64 + q * 16 + lo] + bh[64 + q * 16 + lo];
    biasX[q] = bx[128 + q * 16 + lo];
    biasH[q] = bh[128 + q * 16 + lo];
    wo0[q]   = Wo[(q * 16 + lo) * S_N + 0];
  }
  const float bo0 = bo[0];

  for (int iter = 0; iter < NSTEPS; ++iter) {
    // ================= phase 1: 5 GRU steps =================
    for (int tl = 0; tl < K_N; ++tl) {
      // x_t -> xbuf (bf16)
      {
        const int t = iter + tl;
        for (int i = l; i < 16 * S_N; i += 32) {
          int r = i / S_N, s = i % S_N, b = b0 + r;
          float xv = (b < B_N) ? spk[(b * T_N + t) * S_N + s] : 0.0f;
          xbuf[w][r][s] = (bf16_t)xv;
        }
      }
      // h -> hbuf (bf16) for A-fragment construction
#pragma unroll
      for (int q = 0; q < 4; ++q)
#pragma unroll
        for (int v = 0; v < 8; ++v)
          hbuf[w][v + 8 * hi][q * 16 + lo] = (bf16_t)hreg[q][v];

      // A fragments (16-bit A 16x32: lane=row lo, K contiguous hi*16..hi*16+15)
      v16bf ax, ah0, ah1;
      {
        ax = *(const v16bf*)&xbuf[w][lo][hi * 16];
        ah0 = *(const v16bf*)&hbuf[w][lo][hi * 16];
        ah1 = *(const v16bf*)&hbuf[w][lo][32 + hi * 16];
      }

      // per-q pipeline: r (tile q), z (tile q+4), n (tile q+8, x/h split)
#pragma unroll
      for (int q = 0; q < 4; ++q) {
        v8f cr;
#pragma unroll
        for (int e = 0; e < 8; ++e) cr[e] = biasR[q];
        cr = wmma_bf16(ax,  *(const v16bf*)&sWxT[(q * 16 + lo) * 32 + hi * 16], cr);
        cr = wmma_bf16(ah0, *(const v16bf*)&sWhT[(q * 16 + lo) * 64 + hi * 16], cr);
        cr = wmma_bf16(ah1, *(const v16bf*)&sWhT[(q * 16 + lo) * 64 + 32 + hi * 16], cr);

        const int tz = q + 4;
        v8f cz;
#pragma unroll
        for (int e = 0; e < 8; ++e) cz[e] = biasZ[q];
        cz = wmma_bf16(ax,  *(const v16bf*)&sWxT[(tz * 16 + lo) * 32 + hi * 16], cz);
        cz = wmma_bf16(ah0, *(const v16bf*)&sWhT[(tz * 16 + lo) * 64 + hi * 16], cz);
        cz = wmma_bf16(ah1, *(const v16bf*)&sWhT[(tz * 16 + lo) * 64 + 32 + hi * 16], cz);

        const int tn = q + 8;
        v8f cxn;
#pragma unroll
        for (int e = 0; e < 8; ++e) cxn[e] = biasX[q];
        cxn = wmma_bf16(ax, *(const v16bf*)&sWxT[(tn * 16 + lo) * 32 + hi * 16], cxn);

        v8f chn;
#pragma unroll
        for (int e = 0; e < 8; ++e) chn[e] = biasH[q];
        chn = wmma_bf16(ah0, *(const v16bf*)&sWhT[(tn * 16 + lo) * 64 + hi * 16], chn);
        chn = wmma_bf16(ah1, *(const v16bf*)&sWhT[(tn * 16 + lo) * 64 + 32 + hi * 16], chn);

#pragma unroll
        for (int e = 0; e < 8; ++e) {
          float r = fast_sigmoid(cr[e]);
          float z = fast_sigmoid(cz[e]);
          float n = fast_tanh(cxn[e] + r * chn[e]);
          hreg[q][e] = (1.0f - z) * n + z * hreg[q][e];
        }
      }
    }

    // readout: pred[b] = bo[0] + sum_j h[b][j] * Wo[j][0]
#pragma unroll
    for (int v = 0; v < 8; ++v) {
      float p = 0.0f;
#pragma unroll
      for (int q = 0; q < 4; ++q) p += hreg[q][v] * wo0[q];
      predbuf[w][v + 8 * hi][lo] = p;
    }

    if (l < 16) {
      const int b = b0 + l;
      float p = bo0;
#pragma unroll
      for (int c = 0; c < 16; ++c) p += predbuf[w][l][c];
      if (b < B_N) {
        maskv[b] = (iter > 0 && p > THRESH && prevPred < THRESH) ? 1.0f : 0.0f;
        preds[iter * B_N + b] = p;
      }
      prevPred = p;
    }

    grid_barrier(bar_cnt, bar_gen);   // mask ready, spikes reads of phase 1 done

    // ================= phase 2: event-sparse feedback =================
    // deterministic compaction of firing indices: per-thread chunk of 11,
    // Hillis-Steele prefix scan, ordered scatter into s_list.
    {
      const int c0 = tid * 11;         // 128*11 = 1408 >= 1300
      int cnt = 0;
      for (int k = 0; k < 11; ++k) {
        int c = c0 + k;
        if (c < B_N && maskv[c] != 0.0f) ++cnt;
      }
      s_cnt[tid] = cnt;
      __syncthreads();
      for (int off = 1; off < 128; off <<= 1) {
        int v = s_cnt[tid];
        int add = (tid >= off) ? s_cnt[tid - off] : 0;
        __syncthreads();
        s_cnt[tid] = v + add;
        __syncthreads();
      }
      int pos = s_cnt[tid] - cnt;      // exclusive offset
      const int nf = s_cnt[127];
      for (int k = 0; k < 11; ++k) {
        int c = c0 + k;
        if (c < B_N && maskv[c] != 0.0f) s_list[pos++] = c;
      }
      __syncthreads();

      if (nf > 0) {
        const int gtid = blockIdx.x * 128 + tid;
        for (int row = gtid; row < B_N * S_N; row += NTHR) {
          const float* wrow = wmat + (size_t)row * B_N;   // (b*12+s) row
          float delta = 0.0f;
          for (int k = 0; k < nf; ++k) {
            float wv = wrow[s_list[k]];
            delta += fminf(fmaxf(wv, 0.0f), 10.0f);       // clip(0,10)
          }
          const int b = row / S_N, s = row % S_N;
          spk[(b * T_N + (K_N + iter + 1)) * S_N + s] += delta;
        }
      }
      __syncthreads();                 // s_list/s_cnt reuse safety
    }

    grid_barrier(bar_cnt, bar_gen);    // spike updates visible before next iter
  }
}

// ---------------------------------------------------------------- driver
extern "C" void kernel_launch(void* const* d_in, const int* in_sizes, int n_in,
                              void* d_out, int out_size, void* d_ws, size_t ws_size,
                              hipStream_t stream) {
  const float* spk_in = (const float*)d_in[0];
  const float* wmat   = (const float*)d_in[1];
  const float* Wx     = (const float*)d_in[2];
  const float* Wh     = (const float*)d_in[3];
  const float* bx     = (const float*)d_in[4];
  const float* bh     = (const float*)d_in[5];
  const float* Wo     = (const float*)d_in[6];
  const float* bo     = (const float*)d_in[7];
  float* preds = (float*)d_out;

  char* ws = (char*)d_ws;
  float*    spk   = (float*)(ws + OFF_SPK);
  float*    maskv = (float*)(ws + OFF_MASK);
  unsigned* bar   = (unsigned*)(ws + OFF_BAR);
  bf16_t*   whT   = (bf16_t*)(ws + OFF_WHT);
  bf16_t*   wxT   = (bf16_t*)(ws + OFF_WXT);

  init_kernel<<<512, 256, 0, stream>>>(spk_in, spk, maskv, bar);
  prep_kernel<<<48, 256, 0, stream>>>(Wx, Wh, wxT, whT);
  snn_persistent_kernel<<<NBLK, 128, 0, stream>>>(spk, maskv, bar, wmat, whT, wxT,
                                                  bx, bh, Wo, bo, preds);
}
